// NTM_33389075759105
// MI455X (gfx1250) — matrix operand order repaired and checked by reading, not compiled, not run on hardware
//
#include <hip/hip_runtime.h>
#include <hip/hip_bf16.h>

// Problem constants
#define BB 256
#define TT 128
#define II 64
#define HH 512
#define NN 128
#define MD 64
#define OO 64
#define PD 336      // padded fused projection width (332 valid, padded to 16x)
#define EPSF 1e-8f

#define NT_P (PD/16)   // 21 n-tiles, projection GEMM
#define NT_C (HH/16)   // 32 n-tiles, controller GEMM
#define KT_P (HH/32)   // 16 k-steps, projection GEMM
#define KT_C ((II+MD)/32) // 4 k-steps, controller GEMM

// Column offsets inside the fused projection (raw pre-activations)
#define C_RK 0      // read head: k(64), beta(1), g(1), s(3), y(1) -> [0,70)
#define C_WK 70     // write head: same layout -> [70,140)
#define C_E  140    // erase vec (64)
#define C_A  204    // add vec (64)
#define C_O  268    // output (64) -> [268,332)

typedef _Float16 v16h __attribute__((ext_vector_type(16)));
typedef float    v8f  __attribute__((ext_vector_type(8)));

// A-fragment swizzle within a 32-wide K slab (contiguous k order, same as the B packing):
//   half = k>>4 selects lane group, e = k&15
__device__ __forceinline__ int a_half(int kq) { return (kq >> 4) & 1; }
__device__ __forceinline__ int a_elem(int kq) { return kq & 15; }

// ---------------------------------------------------------------------------
// Fused projection weight, pre-swizzled into WMMA B-fragment order:
// Wsw[((kt*NT_P + nt)*32 + lane)*16 + e] = Wcat[k, c]
//   with k = kt*32 + e + 16*(lane>>4), c = nt*16 + (lane&15)
// ---------------------------------------------------------------------------
__device__ __forceinline__ void wcat_map(int k, int c,
    const float* rWk, const float* rWb, const float* rWg, const float* rWs, const float* rWy,
    const float* rbk, const float* rbb, const float* rbg, const float* rbs, const float* rby,
    const float* wWk, const float* wWb, const float* wWg, const float* wWs, const float* wWy,
    const float* wWe, const float* wWa,
    const float* wbk, const float* wbb, const float* wbg, const float* wbs, const float* wby,
    const float* wbe, const float* wba,
    const float* oW, const float* ob, float* v, float* bv)
{
    float vv = 0.f, bb = 0.f;
    if      (c < 64)   { vv = rWk[k*64 + c];         bb = rbk[c]; }
    else if (c == 64)  { vv = rWb[k];                bb = rbb[0]; }
    else if (c == 65)  { vv = rWg[k];                bb = rbg[0]; }
    else if (c < 69)   { vv = rWs[k*3 + (c-66)];     bb = rbs[c-66]; }
    else if (c == 69)  { vv = rWy[k];                bb = rby[0]; }
    else if (c < 134)  { vv = wWk[k*64 + (c-70)];    bb = wbk[c-70]; }
    else if (c == 134) { vv = wWb[k];                bb = wbb[0]; }
    else if (c == 135) { vv = wWg[k];                bb = wbg[0]; }
    else if (c < 139)  { vv = wWs[k*3 + (c-136)];    bb = wbs[c-136]; }
    else if (c == 139) { vv = wWy[k];                bb = wby[0]; }
    else if (c < 204)  { vv = wWe[k*64 + (c-140)];   bb = wbe[c-140]; }
    else if (c < 268)  { vv = wWa[k*64 + (c-204)];   bb = wba[c-204]; }
    else if (c < 332)  { vv = oW[k*64 + (c-268)];    bb = ob[c-268]; }
    *v = vv; *bv = bb;
}

__global__ void prep_wcat(const float* __restrict__ rWk, const float* __restrict__ rWb,
                          const float* __restrict__ rWg, const float* __restrict__ rWs,
                          const float* __restrict__ rWy,
                          const float* __restrict__ rbk, const float* __restrict__ rbb,
                          const float* __restrict__ rbg, const float* __restrict__ rbs,
                          const float* __restrict__ rby,
                          const float* __restrict__ wWk, const float* __restrict__ wWb,
                          const float* __restrict__ wWg, const float* __restrict__ wWs,
                          const float* __restrict__ wWy, const float* __restrict__ wWe,
                          const float* __restrict__ wWa,
                          const float* __restrict__ wbk, const float* __restrict__ wbb,
                          const float* __restrict__ wbg, const float* __restrict__ wbs,
                          const float* __restrict__ wby, const float* __restrict__ wbe,
                          const float* __restrict__ wba,
                          const float* __restrict__ oW, const float* __restrict__ ob,
                          _Float16* __restrict__ Wsw, float* __restrict__ bcat)
{
    int idx = blockIdx.x * blockDim.x + threadIdx.x;
    const int total = KT_P * NT_P * 512;          // 172032 fragment halves
    for (int i = idx; i < total; i += gridDim.x * blockDim.x) {
        int frag = i >> 9, within = i & 511;
        int lane = within >> 4, e = within & 15;
        int kt = frag / NT_P, nt = frag % NT_P;
        int k = kt * 32 + e + ((lane >> 4) << 4);
        int c = nt * 16 + (lane & 15);
        float v, bv;
        wcat_map(k, c, rWk,rWb,rWg,rWs,rWy, rbk,rbb,rbg,rbs,rby,
                 wWk,wWb,wWg,wWs,wWy,wWe,wWa, wbk,wbb,wbg,wbs,wby,wbe,wba,
                 oW, ob, &v, &bv);
        Wsw[i] = (_Float16)v;
    }
    for (int c = idx; c < PD; c += gridDim.x * blockDim.x) {
        float v, bv;
        wcat_map(0, c, rWk,rWb,rWg,rWs,rWy, rbk,rbb,rbg,rbs,rby,
                 wWk,wWb,wWg,wWs,wWy,wWe,wWa, wbk,wbb,wbg,wbs,wby,wbe,wba,
                 oW, ob, &v, &bv);
        bcat[c] = bv;
    }
}

// Controller weight [W_x ; W_r] (128 x 512), pre-swizzled the same way
__global__ void prep_wctrl(const float* __restrict__ Wx, const float* __restrict__ Wr,
                           const float* __restrict__ cb,
                           _Float16* __restrict__ Wsw, float* __restrict__ bctrl)
{
    int idx = blockIdx.x * blockDim.x + threadIdx.x;
    const int total = KT_C * NT_C * 512;          // 65536 fragment halves
    for (int i = idx; i < total; i += gridDim.x * blockDim.x) {
        int frag = i >> 9, within = i & 511;
        int lane = within >> 4, e = within & 15;
        int kt = frag / NT_C, nt = frag % NT_C;
        int k = kt * 32 + e + ((lane >> 4) << 4);
        int n = nt * 16 + (lane & 15);
        float v = (k < II) ? Wx[k*HH + n] : Wr[(k-II)*HH + n];
        Wsw[i] = (_Float16)v;
    }
    for (int n = idx; n < HH; n += gridDim.x * blockDim.x) bctrl[n] = cb[n];
}

__global__ void init_state(_Float16* __restrict__ h16, float* __restrict__ wr,
                           float* __restrict__ ww, float* __restrict__ Mg)
{
    int gid = blockIdx.x * blockDim.x + threadIdx.x;
    int stride = gridDim.x * blockDim.x;
    for (int i = gid; i < BB*HH; i += stride) h16[i] = (_Float16)0.f;
    for (int i = gid; i < BB*NN; i += stride) {
        float v = (i == 0) ? 1.f : 0.f;   // reference: only [batch0, slot0] = 1
        wr[i] = v; ww[i] = v;
    }
    for (int i = gid; i < BB*NN*MD; i += stride) Mg[i] = 1e-6f;
}

// ---------------------------------------------------------------------------
// Fused projection GEMM: proj[256,336] = h16[256,512] @ Wcat + bcat  (raw).
// One block = one 16-row m-tile staged once into LDS in A-fragment order;
// 8 waves cover 8 n-tiles (3 block-groups cover all 21).
// ---------------------------------------------------------------------------
__global__ void __launch_bounds__(256)
gemm_proj(const _Float16* __restrict__ h16,
          const _Float16* __restrict__ Wsw,
          const float* __restrict__ bcat,
          float* __restrict__ proj,
          float* __restrict__ outp, int t)
{
    __shared__ __align__(16) _Float16 Ash[16 * HH];   // 16 KB, fragment order
    int tid = threadIdx.x;
    int mt = blockIdx.x / 3, g = blockIdx.x % 3;
    int m0 = mt * 16;

    // stage A tile: coalesced u32 global reads -> fragment-order LDS stores
    {
        const unsigned* hsrc = (const unsigned*)(h16 + (size_t)m0 * HH);
        unsigned* Ash32 = (unsigned*)Ash;
        for (int i = tid; i < 16 * HH / 2; i += 256) {
            int m = i >> 8;               // 256 u32 per 512-half row
            int k = (i & 255) * 2;        // even k
            int kt = k >> 5, kq = k & 31;
            int pos = kt * 512 + ((a_half(kq) << 4) + m) * 16 + a_elem(kq);
            Ash32[pos >> 1] = hsrc[i];    // even e -> u32-aligned
        }
    }
    __syncthreads();

    int wave = tid >> 5, lane = tid & 31;
    int nt = g * 8 + wave;
    if (nt >= NT_P) return;
    int half = lane >> 4, lm = lane & 15;
    int n0 = nt * 16;

    v8f c = {};
#pragma unroll 4
    for (int kt = 0; kt < KT_P; ++kt) {
        v16h a = *(const v16h*)&Ash[kt * 512 + lane * 16];
        const v16h* bp = (const v16h*)&Wsw[(((size_t)kt * NT_P + nt) * 32 + lane) * 16];
        if (kt + 1 < KT_P)
            __builtin_prefetch(&Wsw[(((size_t)(kt+1) * NT_P + nt) * 32 + lane) * 16], 0, 1);
        c = __builtin_amdgcn_wmma_f32_16x16x32_f16(false, a, false, *bp,
                                                   (short)0, c, false, false);
    }
    int n = n0 + lm;
    float bv = bcat[n];
#pragma unroll
    for (int rr = 0; rr < 8; ++rr) {
        int m = m0 + rr + half * 8;
        float v = c[rr] + bv;
        proj[m * PD + n] = v;
        if (outp && n >= C_O && n < C_O + OO)
            outp[((size_t)t * BB + m) * OO + (n - C_O)] = v;
    }
}

// ---------------------------------------------------------------------------
// Controller GEMM: h16 = f16(tanh([x_t | r][256,128] @ Wctrl[128,512] + b)).
// Same structure; A tile (16x128) converted f32->f16 during LDS staging.
// ---------------------------------------------------------------------------
__global__ void __launch_bounds__(256)
gemm_ctrl(const float* __restrict__ x, int t,
          const float* __restrict__ rbuf,
          const _Float16* __restrict__ Wsw,
          const float* __restrict__ bctrl,
          _Float16* __restrict__ h16)
{
    __shared__ __align__(16) _Float16 Ash[16 * (II + MD)];   // 4 KB
    int tid = threadIdx.x;
    int mt = blockIdx.x / 4, g = blockIdx.x % 4;
    int m0 = mt * 16;

    // stage [x_t | r] tile, f32 -> f16, fragment-order
    for (int i = tid; i < 16 * (II + MD) / 2; i += 256) {
        int m = i >> 6;                  // 64 half-pairs per 128-half row
        int k = (i & 63) * 2;            // even k; region boundary (64) is even
        int row = m0 + m;
        float v0, v1;
        if (k < II) {
            const float* xp = x + ((size_t)row * TT + t) * II;
            v0 = xp[k]; v1 = xp[k + 1];
        } else {
            const float* rp = rbuf + (size_t)row * MD + (k - II);
            v0 = rp[0]; v1 = rp[1];
        }
        int kt = k >> 5, kq = k & 31;
        int pos = kt * 512 + ((a_half(kq) << 4) + m) * 16 + a_elem(kq);
        Ash[pos]     = (_Float16)v0;
        Ash[pos + 1] = (_Float16)v1;
    }
    __syncthreads();

    int wave = tid >> 5, lane = tid & 31;
    int nt = g * 8 + wave;               // 32 n-tiles, all valid
    int half = lane >> 4, lm = lane & 31 & 15;
    int n0 = nt * 16;

    v8f c = {};
#pragma unroll
    for (int kt = 0; kt < KT_C; ++kt) {
        v16h a = *(const v16h*)&Ash[kt * 512 + lane * 16];
        const v16h* bp = (const v16h*)&Wsw[(((size_t)kt * NT_C + nt) * 32 + lane) * 16];
        c = __builtin_amdgcn_wmma_f32_16x16x32_f16(false, a, false, *bp,
                                                   (short)0, c, false, false);
    }
    int n = n0 + lm;
    float bv = bctrl[n];
#pragma unroll
    for (int rr = 0; rr < 8; ++rr) {
        int m = m0 + rr + half * 8;
        h16[m * HH + n] = (_Float16)tanhf(c[rr] + bv);
    }
}

// ---------------------------------------------------------------------------
// NTM addressing (one workgroup of 128 threads = 4 waves, one batch element,
// memory slice staged in LDS).
// ---------------------------------------------------------------------------
__device__ __forceinline__ float reduceN(float v, float* red, int tid, int domax)
{
    red[tid] = v;
    __syncthreads();
    for (int s = NN / 2; s > 0; s >>= 1) {
        if (tid < s) {
            float a = red[tid], b = red[tid + s];
            red[tid] = domax ? fmaxf(a, b) : (a + b);
        }
        __syncthreads();
    }
    float r = red[0];
    __syncthreads();
    return r;
}

__device__ __forceinline__ float softplusf(float x)
{
    return (x > 20.f) ? x : log1pf(expf(x));
}

// ph: head block of raw projections (k[0..64), beta@64, g@65, s@66..68, y@69)
__device__ void ntm_address(const float* __restrict__ ph,
                            const float* __restrict__ wprev,   // [NN] global carry
                            const float* __restrict__ Msh,     // [NN*MD] LDS
                            float* ksh, float* wbuf, float* red, float* scal,
                            float* wout, int tid)
{
    if (tid < MD) ksh[tid] = tanhf(ph[tid]);
    if (tid == MD) {
        scal[0] = softplusf(ph[64]);                 // beta
        scal[1] = 1.f / (1.f + expf(-ph[65]));       // g
        float s0 = ph[66], s1 = ph[67], s2 = ph[68];
        float mx = fmaxf(s0, fmaxf(s1, s2));
        float e0 = expf(s0 - mx), e1 = expf(s1 - mx), e2 = expf(s2 - mx);
        float ss = e0 + e1 + e2;
        scal[2] = e0 / ss; scal[3] = e1 / ss; scal[4] = e2 / ss;
        scal[5] = 1.f + softplusf(ph[69]);           // gamma
    }
    __syncthreads();
    if (tid == 0) {
        float kn = 0.f;
        for (int m = 0; m < MD; ++m) kn += ksh[m] * ksh[m];
        scal[6] = sqrtf(kn);
    }
    __syncthreads();
    float beta = scal[0], g = scal[1];
    float s0 = scal[2], s1 = scal[3], s2 = scal[4];
    float gamma = scal[5], kn = scal[6];

    const float* row = Msh + tid * MD;
    float dot = 0.f, rn = 0.f;
    for (int m = 0; m < MD; ++m) { float mv = row[m]; dot += mv * ksh[m]; rn += mv * mv; }
    float sim = dot / (sqrtf(rn) * kn + EPSF);
    float sc = beta * sim;
    float mx = reduceN(sc, red, tid, 1);
    float ex = expf(sc - mx);
    float ssum = reduceN(ex, red, tid, 0);
    float wc = ex / ssum;
    float wg = g * wc + (1.f - g) * wprev[tid];
    wbuf[tid] = wg;
    __syncthreads();
    float ws = s0 * wbuf[(tid + 1) & (NN - 1)] + s1 * wg + s2 * wbuf[(tid + NN - 1) & (NN - 1)];
    float wp = powf(ws + EPSF, gamma);
    float psum = reduceN(wp, red, tid, 0);
    wout[tid] = wp / psum;
    __syncthreads();
}

// write-address + erase/add memory update + read-address + read vector
__global__ void __launch_bounds__(NN)
mem_step(const float* __restrict__ proj, float* __restrict__ Mg,
         const float* __restrict__ wr, const float* __restrict__ ww,
         float* __restrict__ rbuf)
{
    __shared__ float Msh[NN * MD];
    __shared__ float ksh[MD], evec[MD], avec[MD];
    __shared__ float wbuf[NN], red[NN], waddr[NN];
    __shared__ float scal[8];
    int b = blockIdx.x, tid = threadIdx.x;

    float4* Ms4 = (float4*)Msh;
    const float4* Mg4 = (const float4*)(Mg + (size_t)b * NN * MD);
    for (int i = tid; i < NN * MD / 4; i += NN) Ms4[i] = Mg4[i];
    __syncthreads();

    const float* pb = proj + (size_t)b * PD;
    // write head address (prev h, prev ww carry, old M)
    ntm_address(pb + C_WK, ww + (size_t)b * NN, Msh, ksh, wbuf, red, scal, waddr, tid);
    if (tid < MD) {
        evec[tid] = 1.f / (1.f + expf(-pb[C_E + tid]));
        avec[tid] = tanhf(pb[C_A + tid]);
    }
    __syncthreads();
    {   // erase/add on row tid
        float w = waddr[tid];
        float* row = Msh + tid * MD;
        for (int m = 0; m < MD; ++m)
            row[m] = row[m] * (1.f - w * evec[m]) + w * avec[m];
    }
    __syncthreads();
    float4* dst4 = (float4*)(Mg + (size_t)b * NN * MD);
    for (int i = tid; i < NN * MD / 4; i += NN) dst4[i] = Ms4[i];

    // read head address on the updated memory, then read vector
    ntm_address(pb + C_RK, wr + (size_t)b * NN, Msh, ksh, wbuf, red, scal, waddr, tid);
    if (tid < MD) {
        float acc = 0.f;
        for (int n = 0; n < NN; ++n) acc += waddr[n] * Msh[n * MD + tid];
        rbuf[(size_t)b * MD + tid] = acc;
    }
}

// recompute carry weights with the NEW hidden state's projections
__global__ void __launch_bounds__(NN)
addr2(const float* __restrict__ proj, const float* __restrict__ Mg,
      float* __restrict__ wr, float* __restrict__ ww)
{
    __shared__ float Msh[NN * MD];
    __shared__ float ksh[MD];
    __shared__ float wbuf[NN], red[NN], waddr[NN];
    __shared__ float scal[8];
    int b = blockIdx.x, tid = threadIdx.x;

    float4* Ms4 = (float4*)Msh;
    const float4* Mg4 = (const float4*)(Mg + (size_t)b * NN * MD);
    for (int i = tid; i < NN * MD / 4; i += NN) Ms4[i] = Mg4[i];
    __syncthreads();

    const float* pb = proj + (size_t)b * PD;
    ntm_address(pb + C_RK, wr + (size_t)b * NN, Msh, ksh, wbuf, red, scal, waddr, tid);
    float newr = waddr[tid];
    __syncthreads();
    ntm_address(pb + C_WK, ww + (size_t)b * NN, Msh, ksh, wbuf, red, scal, waddr, tid);
    float neww = waddr[tid];
    wr[(size_t)b * NN + tid] = newr;
    ww[(size_t)b * NN + tid] = neww;
}

// ---------------------------------------------------------------------------
extern "C" void kernel_launch(void* const* d_in, const int* in_sizes, int n_in,
                              void* d_out, int out_size, void* d_ws, size_t ws_size,
                              hipStream_t stream)
{
    // Input order: jax pytree (sorted dict keys): ctrl{W_r,W_x,b}, out{W,b},
    // read{W_b,W_g,W_k,W_s,W_y,b_b,b_g,b_k,b_s,b_y},
    // write{W_a,W_b,W_e,W_g,W_k,W_s,W_y,b_a,b_b,b_e,b_g,b_k,b_s,b_y}, x
    const float* cWr = (const float*)d_in[0];
    const float* cWx = (const float*)d_in[1];
    const float* cb  = (const float*)d_in[2];
    const float* oW  = (const float*)d_in[3];
    const float* ob  = (const float*)d_in[4];
    const float* rWb = (const float*)d_in[5];
    const float* rWg = (const float*)d_in[6];
    const float* rWk = (const float*)d_in[7];
    const float* rWs = (const float*)d_in[8];
    const float* rWy = (const float*)d_in[9];
    const float* rbb = (const float*)d_in[10];
    const float* rbg = (const float*)d_in[11];
    const float* rbk = (const float*)d_in[12];
    const float* rbs = (const float*)d_in[13];
    const float* rby = (const float*)d_in[14];
    const float* wWa = (const float*)d_in[15];
    const float* wWb = (const float*)d_in[16];
    const float* wWe = (const float*)d_in[17];
    const float* wWg = (const float*)d_in[18];
    const float* wWk = (const float*)d_in[19];
    const float* wWs = (const float*)d_in[20];
    const float* wWy = (const float*)d_in[21];
    const float* wba = (const float*)d_in[22];
    const float* wbb = (const float*)d_in[23];
    const float* wbe = (const float*)d_in[24];
    const float* wbg = (const float*)d_in[25];
    const float* wbk = (const float*)d_in[26];
    const float* wbs = (const float*)d_in[27];
    const float* wby = (const float*)d_in[28];
    const float* x   = (const float*)d_in[29];
    float* out = (float*)d_out;

    // Workspace carve-out (all fits in L2)
    char* ws = (char*)d_ws;
    size_t off = 0;
    auto carve = [&](size_t bytes) {
        void* p = ws + off;
        off = (off + bytes + 255) & ~(size_t)255;
        return p;
    };
    _Float16* h16   = (_Float16*)carve(sizeof(_Float16) * BB * HH);
    float*    proj  = (float*)   carve(sizeof(float) * BB * PD);
    float*    Mg    = (float*)   carve(sizeof(float) * BB * NN * MD);
    float*    wrb   = (float*)   carve(sizeof(float) * BB * NN);
    float*    wwb   = (float*)   carve(sizeof(float) * BB * NN);
    float*    rbuf  = (float*)   carve(sizeof(float) * BB * MD);
    _Float16* Wcat  = (_Float16*)carve(sizeof(_Float16) * KT_P * NT_P * 512);
    float*    bcat  = (float*)   carve(sizeof(float) * PD);
    _Float16* Wctrl = (_Float16*)carve(sizeof(_Float16) * KT_C * NT_C * 512);
    float*    bctrl = (float*)   carve(sizeof(float) * HH);
    (void)ws_size; (void)in_sizes; (void)n_in; (void)out_size;

    prep_wcat<<<672, 256, 0, stream>>>(rWk, rWb, rWg, rWs, rWy,
                                       rbk, rbb, rbg, rbs, rby,
                                       wWk, wWb, wWg, wWs, wWy, wWe, wWa,
                                       wbk, wbb, wbg, wbs, wby, wbe, wba,
                                       oW, ob, Wcat, bcat);
    prep_wctrl<<<256, 256, 0, stream>>>(cWx, cWr, cb, Wctrl, bctrl);
    init_state<<<1024, 256, 0, stream>>>(h16, wrb, wwb, Mg);

    // projections of h0 (= biases): A is all zeros, GEMM yields bcat
    gemm_proj<<<48, 256, 0, stream>>>(h16, Wcat, bcat, proj, nullptr, 0);

    for (int t = 0; t < TT; ++t) {
        mem_step<<<BB, NN, 0, stream>>>(proj, Mg, wrb, wwb, rbuf);
        gemm_ctrl<<<64, 256, 0, stream>>>(x, t, rbuf, Wctrl, bctrl, h16);
        gemm_proj<<<48, 256, 0, stream>>>(h16, Wcat, bcat, proj, out, t);
        addr2<<<BB, NN, 0, stream>>>(proj, Mg, wrb, wwb);
    }
}
